// UnitGCN_52974126628952
// MI455X (gfx1250) — matrix operand order repaired and checked by reading, not compiled, not run on hardware
//
#include <hip/hip_runtime.h>

// ---------------- problem constants (from reference) ----------------
#define K_SUB   3
#define V_J     25
#define C_IN    64
#define C_OUT   128
#define T_DIM   128
#define N_BATCH 64
#define TT      16                    // t values per block
#define COLS    (TT * V_J)            // 400 GEMM columns per block
#define NTILES  (COLS / 16)           // 25 column tiles of 16
#define TV      (T_DIM * V_J)         // 3200
#define NBLK    ((T_DIM / TT) * N_BATCH)   // 512 blocks in kernel 1
#define BN_EPS  1e-5f

#define XP_STRIDE  72                 // halves per Xp row (pad 64 -> 72)
#define YS_STRIDE  20                 // floats per AccR row (pad 16 -> 20)
#define YSB_STRIDE 40                 // halves per Ysb (m,t) row (pad 32 -> 40)
#define ABF_STRIDE 40                 // halves per Abf (k,w) row (pad 32 -> 40)

#define XP_BYTES   (COLS * XP_STRIDE * 2)            // 57,600
#define ABF_BYTES  (K_SUB * 32 * ABF_STRIDE * 2)     //  7,680
#define YSB_BYTES  (16 * 16 * YSB_STRIDE * 2)        // 20,480
#define POOL_BYTES (XP_BYTES + ABF_BYTES)            // 65,280 (>= 52,480)

typedef __attribute__((ext_vector_type(16))) __bf16 v16bf;
typedef __attribute__((ext_vector_type(2)))  __bf16 v2bf;
typedef __attribute__((ext_vector_type(8)))  float  v8f;

union FragAB { v16bf v; uint4 q[2]; unsigned short u[16]; };

__device__ __forceinline__ unsigned short f2bf1(float f) {
#if __has_builtin(__builtin_amdgcn_cvt_pk_bf16_f32)
  union { v2bf v; unsigned short u[2]; } r;
  r.v = __builtin_amdgcn_cvt_pk_bf16_f32(f, 0.0f);   // v_cvt_pk_bf16_f32
  return r.u[0];
#else
  union { float f; unsigned int u; } x; x.f = f;
  unsigned int u = x.u;
  u += 0x7FFFu + ((u >> 16) & 1u);       // RTNE truncation fallback
  return (unsigned short)(u >> 16);
#endif
}

__device__ __forceinline__ unsigned int f2bf_pk(float lo, float hi) {
#if __has_builtin(__builtin_amdgcn_cvt_pk_bf16_f32)
  union { v2bf v; unsigned int u; } r;
  r.v = __builtin_amdgcn_cvt_pk_bf16_f32(lo, hi);
  return r.u;
#else
  return (unsigned int)f2bf1(lo) | ((unsigned int)f2bf1(hi) << 16);
#endif
}

__device__ __forceinline__ float bf2f(unsigned short h) {
  union { unsigned int u; float f; } x;
  x.u = (unsigned int)h << 16;
  return x.f;
}

// =====================================================================
// Kernel 0: pack W = [Wc(384x64); Wd(128x64)] as bf16, K-sequential rows.
// =====================================================================
__global__ __launch_bounds__(256)
void w_pack_kernel(const float* __restrict__ Wc, const float* __restrict__ Wd,
                   unsigned short* __restrict__ Wb) {
  int i = blockIdx.x * 256 + threadIdx.x;          // 0 .. 512*64-1
  int row = i >> 6, k = i & 63;
  float v = (row < K_SUB * C_OUT) ? Wc[row * C_IN + k]
                                  : Wd[(row - K_SUB * C_OUT) * C_IN + k];
  Wb[i] = f2bf1(v);
}

// =====================================================================
// Kernel 1: all-WMMA pipeline (main conv, joint aggregation, residual)
// + register-side deterministic BN partial sums per block -> pbuf.
// Intermediates stored bf16 (pre-BN) to halve HBM traffic.
// Block = (n, t-slab of 16). 256 threads = 8 wave32.
// =====================================================================
__global__ __launch_bounds__(256)
void gcn_wmma_kernel(const float* __restrict__ x,
                     const float* __restrict__ A,
                     const unsigned short* __restrict__ Wb,
                     const float* __restrict__ bc,
                     const float* __restrict__ bd,
                     unsigned short* __restrict__ ybuf,
                     unsigned short* __restrict__ rbuf,
                     float* __restrict__ pbuf) {
  // pool phase A: Xp[COLS][72] bf16 | Abf[3][32][40] bf16  (65,280 B)
  // pool phase B: Ysb[16][16][40] bf16 | AccR[400][20] f32 (52,480 B)
  //               (Abf region sits above 52,480 -> survives phase B)
  __shared__ __align__(16) char pool[POOL_BYTES];
  __shared__ __align__(32) float bcs[C_OUT * K_SUB];
  __shared__ __align__(32) float bds[C_OUT];
  __shared__ float psY[C_OUT], psY2[C_OUT], psR[C_OUT], psR2[C_OUT];
  __shared__ float redS[8][16], redQ[8][16];

  unsigned short* XpU  = (unsigned short*)pool;               // [col][c]
  unsigned short* AbfU = (unsigned short*)(pool + XP_BYTES);  // [k][w][v]
  unsigned short* YsbU = (unsigned short*)pool;               // [m][t][v]
  float (*AccR)[YS_STRIDE] = (float (*)[YS_STRIDE])(pool + YSB_BYTES);

  const int tg    = blockIdx.x;
  const int n     = blockIdx.y;
  const int t0    = tg * TT;
  const int tid   = threadIdx.x;
  const int lane  = tid & 31;
  const int half  = lane >> 4;
  const int ln16  = lane & 15;
  const int waveu = __builtin_amdgcn_readfirstlane(tid >> 5); // scalar wave id

  // ---- phase A: stage X transposed bf16 + padded bf16 adjacency ----
  const float* xbase = x + (size_t)n * C_IN * TV + (size_t)t0 * V_J;
  for (int e = tid; e < C_IN * COLS; e += 256) {
    int c = e / COLS, col = e % COLS;                 // coalesced global read
    XpU[col * XP_STRIDE + c] = f2bf1(xbase[(size_t)c * TV + col]);
  }
  for (int e = tid; e < K_SUB * 32 * ABF_STRIDE; e += 256) {
    int k = e / (32 * ABF_STRIDE), rem = e % (32 * ABF_STRIDE);
    int w = rem / ABF_STRIDE, v = rem % ABF_STRIDE;
    AbfU[e] = (v < V_J && w < V_J) ? f2bf1(A[(k * V_J + v) * V_J + w])
                                   : (unsigned short)0;
  }
  for (int e = tid; e < C_OUT * K_SUB; e += 256) bcs[e] = bc[e];
  for (int e = tid; e < C_OUT; e += 256)         bds[e] = bd[e];
  if (tid < C_OUT) { psY[tid] = 0.f; psY2[tid] = 0.f; psR[tid] = 0.f; psR2[tid] = 0.f; }
  __syncthreads();

  // ---- preload this wave's main-GEMM B fragments (kept all kernel) ----
  FragAB b0f[4], b1f[4];
#pragma unroll
  for (int i = 0; i < 4; ++i) {
    int jt = waveu + 8 * i;                          // scalar
    if (jt < NTILES) {
      int col = jt * 16 + ln16;
      b0f[i].q[0] = *(const uint4*)&XpU[col * XP_STRIDE + half * 8];
      b0f[i].q[1] = *(const uint4*)&XpU[col * XP_STRIDE + 16 + half * 8];
      b1f[i].q[0] = *(const uint4*)&XpU[col * XP_STRIDE + 32 + half * 8];
      b1f[i].q[1] = *(const uint4*)&XpU[col * XP_STRIDE + 48 + half * 8];
    }
  }
  __syncthreads();          // Xp dead; pool becomes Ysb/AccR

  // zero Ysb v-padding (v = 25..31); never overwritten afterwards
  for (int e = tid; e < 256 * 7; e += 256) {
    int row = e / 7, v = V_J + e % 7;
    YsbU[row * YSB_STRIDE + v] = 0;
  }

  v8f aggAcc[2][2] = {};

  for (int c0 = 0; c0 < C_OUT; c0 += 16) {
    // ---------------- main path: 3 k-subsets ----------------
    for (int k = 0; k < K_SUB; ++k) {
      FragAB a0, a1;
      {
        const uint4* wp = (const uint4*)(Wb + (size_t)(k * C_OUT + c0 + ln16) * C_IN);
        a0.q[0] = wp[half];     a0.q[1] = wp[2 + half];
        a1.q[0] = wp[4 + half]; a1.q[1] = wp[6 + half];
      }
      v8f biasv = *(const v8f*)&bcs[k * C_OUT + c0 + 8 * half];
#pragma unroll
      for (int i = 0; i < 4; ++i) {
        int jt = waveu + 8 * i;                      // scalar -> s_cbranch guard
        if (jt < NTILES) {
          v8f acc = {};
          acc = __builtin_amdgcn_wmma_f32_16x16x32_bf16(false, a0.v, false, b0f[i].v,
                                                        (short)0, acc, false, false);
          acc = __builtin_amdgcn_wmma_f32_16x16x32_bf16(false, a1.v, false, b1f[i].v,
                                                        (short)0, acc, false, false);
          v8f vb = acc + biasv;
          int col = jt * 16 + ln16;
          int t = col / V_J, v = col % V_J;
          unsigned short* yb = &YsbU[t * YSB_STRIDE + v];
#pragma unroll
          for (int r = 0; r < 8; ++r)
            yb[(r + 8 * half) * (16 * YSB_STRIDE)] = f2bf1(vb[r]);
        }
      }
      __syncthreads();

      // ---- aggregation WMMA: out[t][w] += sum_v Y[m][t][v] * A_k[v][w] ----
      FragAB ab0, ab1;
      ab0.v = *(const v16bf*)&AbfU[(k * 32 + ln16) * ABF_STRIDE + half * 16];
      ab1.v = *(const v16bf*)&AbfU[(k * 32 + 16 + ln16) * ABF_STRIDE + half * 16];
#pragma unroll
      for (int mi = 0; mi < 2; ++mi) {
        int m = waveu + 8 * mi;                      // scalar
        FragAB ya;
        ya.q[0] = *(const uint4*)&YsbU[(m * 16 + ln16) * YSB_STRIDE + half * 16];
        ya.q[1] = *(const uint4*)&YsbU[(m * 16 + ln16) * YSB_STRIDE + half * 16 + 8];
        aggAcc[mi][0] = __builtin_amdgcn_wmma_f32_16x16x32_bf16(false, ya.v, false, ab0.v,
                                                 (short)0, aggAcc[mi][0], false, false);
        aggAcc[mi][1] = __builtin_amdgcn_wmma_f32_16x16x32_bf16(false, ya.v, false, ab1.v,
                                                 (short)0, aggAcc[mi][1], false, false);
      }
      __syncthreads();     // Ysb consumed before next k overwrites it
    }

    // ---- stage aggregated chunk to AccR ----
#pragma unroll
    for (int mi = 0; mi < 2; ++mi) {
      int m = waveu + 8 * mi;
#pragma unroll
      for (int nt = 0; nt < 2; ++nt) {
        int w = nt * 16 + ln16;
        if (w < V_J) {
#pragma unroll
          for (int r = 0; r < 8; ++r) {
            int t = r + 8 * half;
            AccR[t * V_J + w][m] = aggAcc[mi][nt][r];
          }
        }
      }
    }

    // ---- y-path BN partials straight from registers (wave owns m) ----
#pragma unroll
    for (int mi = 0; mi < 2; ++mi) {
      int m = waveu + 8 * mi;
      float s = 0.f, q = 0.f;
#pragma unroll
      for (int nt = 0; nt < 2; ++nt) {
        int w = nt * 16 + ln16;
        if (w < V_J) {
#pragma unroll
          for (int r = 0; r < 8; ++r) { float a = aggAcc[mi][nt][r]; s += a; q += a * a; }
        }
      }
#pragma unroll
      for (int off = 16; off > 0; off >>= 1) {
        s += __shfl_xor(s, off, 32);
        q += __shfl_xor(q, off, 32);
      }
      if (lane == 0) { psY[c0 + m] += s; psY2[c0 + m] += q; }
      v8f z = {};
      aggAcc[mi][0] = z; aggAcc[mi][1] = z;
    }
    __syncthreads();

    // coalesced ybuf write (bf16, packed pairs -> b32 stores)
    {
      unsigned short* ydst = ybuf + ((size_t)(n * C_OUT + c0) * T_DIM + t0) * V_J;
      for (int e = tid * 2; e < 16 * COLS; e += 512) {
        int c = e / COLS, col = e % COLS;            // col even; pair in same row
        unsigned int pk = f2bf_pk(AccR[col][c], AccR[col + 1][c]);
        *(unsigned int*)&ydst[(size_t)c * TV + col] = pk;
      }
    }
    __syncthreads();   // AccR reads done before residual overwrites it

    // ---------------- residual path: Wb rows 384 + c0 + m ----------------
    {
      FragAB a0, a1;
      const uint4* wp = (const uint4*)(Wb + (size_t)(K_SUB * C_OUT + c0 + ln16) * C_IN);
      a0.q[0] = wp[half];     a0.q[1] = wp[2 + half];
      a1.q[0] = wp[4 + half]; a1.q[1] = wp[6 + half];
      v8f biasv = *(const v8f*)&bds[c0 + 8 * half];
      float rsum[8] = {0.f, 0.f, 0.f, 0.f, 0.f, 0.f, 0.f, 0.f};
      float rqq[8]  = {0.f, 0.f, 0.f, 0.f, 0.f, 0.f, 0.f, 0.f};
#pragma unroll
      for (int i = 0; i < 4; ++i) {
        int jt = waveu + 8 * i;
        if (jt < NTILES) {
          v8f acc = {};
          acc = __builtin_amdgcn_wmma_f32_16x16x32_bf16(false, a0.v, false, b0f[i].v,
                                                        (short)0, acc, false, false);
          acc = __builtin_amdgcn_wmma_f32_16x16x32_bf16(false, a1.v, false, b1f[i].v,
                                                        (short)0, acc, false, false);
          v8f vb = acc + biasv;
          *(v8f*)&AccR[jt * 16 + ln16][8 * half] = vb;
#pragma unroll
          for (int r = 0; r < 8; ++r) { rsum[r] += vb[r]; rqq[r] += vb[r] * vb[r]; }
        }
      }
      // reduce within 16-lane halves; lane0 of each half owns channels r(+8)
#pragma unroll
      for (int r = 0; r < 8; ++r) {
        float s = rsum[r], q = rqq[r];
#pragma unroll
        for (int off = 8; off > 0; off >>= 1) {
          s += __shfl_xor(s, off, 16);
          q += __shfl_xor(q, off, 16);
        }
        if (ln16 == 0) { redS[waveu][8 * half + r] = s; redQ[waveu][8 * half + r] = q; }
      }
    }
    __syncthreads();
    if (tid < 16) {
      float a = 0.f, b2 = 0.f;
      for (int wv = 0; wv < 8; ++wv) { a += redS[wv][tid]; b2 += redQ[wv][tid]; }
      psR[c0 + tid] += a; psR2[c0 + tid] += b2;
    }

    // coalesced rbuf write (bf16, packed pairs -> b32 stores)
    {
      unsigned short* rdst = rbuf + ((size_t)(n * C_OUT + c0) * T_DIM + t0) * V_J;
      for (int e = tid * 2; e < 16 * COLS; e += 512) {
        int c = e / COLS, col = e % COLS;
        unsigned int pk = f2bf_pk(AccR[col][c], AccR[col + 1][c]);
        *(unsigned int*)&rdst[(size_t)c * TV + col] = pk;
      }
    }
    __syncthreads();   // protect AccR before next chunk
  }

  // ---- write per-block partials: [ysum|ysumsq|rsum|rsumsq] x 128 ----
  {
    int b = n * gridDim.x + tg;
    float* pb = pbuf + (size_t)b * (4 * C_OUT);
    if (tid < C_OUT) {
      pb[tid]             = psY[tid];
      pb[C_OUT + tid]     = psY2[tid];
      pb[2 * C_OUT + tid] = psR[tid];
      pb[3 * C_OUT + tid] = psR2[tid];
    }
  }
}

// =====================================================================
// Kernel 2: reduce per-block partials -> per-channel {mean, var}.
// =====================================================================
__global__ __launch_bounds__(256)
void gcn_stats_kernel(const float* __restrict__ pbuf, float* __restrict__ stats) {
  const int ch = blockIdx.x;
  const int c  = ch & (C_OUT - 1);
  const int so = (ch < C_OUT) ? 0 : 2 * C_OUT;
  float s = 0.0f, ss = 0.0f;
  for (int b = threadIdx.x; b < NBLK; b += 256) {
    const float* pb = pbuf + (size_t)b * (4 * C_OUT);
    s  += pb[so + c];
    ss += pb[so + C_OUT + c];
  }
  __shared__ float sh[256], sh2[256];
  sh[threadIdx.x] = s; sh2[threadIdx.x] = ss;
  __syncthreads();
  for (int st = 128; st > 0; st >>= 1) {
    if (threadIdx.x < st) {
      sh[threadIdx.x]  += sh[threadIdx.x + st];
      sh2[threadIdx.x] += sh2[threadIdx.x + st];
    }
    __syncthreads();
  }
  if (threadIdx.x == 0) {
    const float cnt = (float)((long long)N_BATCH * TV);
    float mean = sh[0] / cnt;
    float var  = sh2[0] / cnt - mean * mean;   // biased, like jnp.var
    stats[ch * 2]     = mean;
    stats[ch * 2 + 1] = var;
  }
}

// =====================================================================
// Kernel 3: fused BN(main) + BN(residual) + add + ReLU, 4 elems/thread.
// 4-groups never cross a channel (4 | TV), so stats are uniform per thread.
// =====================================================================
__global__ __launch_bounds__(256)
void gcn_finalize_kernel(const unsigned short* __restrict__ ybuf,
                         const unsigned short* __restrict__ rbuf,
                         const float* __restrict__ stats,
                         const float* __restrict__ gamma,
                         const float* __restrict__ beta,
                         const float* __restrict__ gamma_d,
                         const float* __restrict__ beta_d,
                         float* __restrict__ out, long long total) {
  long long i4 = ((long long)blockIdx.x * 256 + threadIdx.x) * 4;
  if (i4 >= total) return;
  int c = (int)((i4 / TV) % C_OUT);
  float m  = stats[c * 2],           v  = stats[c * 2 + 1];
  float md = stats[(C_OUT + c) * 2], vd = stats[(C_OUT + c) * 2 + 1];
  float sy = rsqrtf(v  + BN_EPS) * gamma[c];
  float sr = rsqrtf(vd + BN_EPS) * gamma_d[c];
  float oy = beta[c]   - m  * sy;
  float orr = beta_d[c] - md * sr;

  uint2 yu = *(const uint2*)&ybuf[i4];
  uint2 ru = *(const uint2*)&rbuf[i4];
  float4 o;
  {
    float y0 = bf2f((unsigned short)(yu.x & 0xFFFF)), r0 = bf2f((unsigned short)(ru.x & 0xFFFF));
    float y1 = bf2f((unsigned short)(yu.x >> 16)),    r1 = bf2f((unsigned short)(ru.x >> 16));
    float y2 = bf2f((unsigned short)(yu.y & 0xFFFF)), r2 = bf2f((unsigned short)(ru.y & 0xFFFF));
    float y3 = bf2f((unsigned short)(yu.y >> 16)),    r3 = bf2f((unsigned short)(ru.y >> 16));
    o.x = y0 * sy + oy + (r0 * sr + orr);
    o.y = y1 * sy + oy + (r1 * sr + orr);
    o.z = y2 * sy + oy + (r2 * sr + orr);
    o.w = y3 * sy + oy + (r3 * sr + orr);
  }
  o.x = o.x > 0.f ? o.x : 0.f;
  o.y = o.y > 0.f ? o.y : 0.f;
  o.z = o.z > 0.f ? o.z : 0.f;
  o.w = o.w > 0.f ? o.w : 0.f;
  *(float4*)&out[i4] = o;
}

// =====================================================================
// launch
// =====================================================================
extern "C" void kernel_launch(void* const* d_in, const int* in_sizes, int n_in,
                              void* d_out, int out_size, void* d_ws, size_t ws_size,
                              hipStream_t stream) {
  const float* x       = (const float*)d_in[0];
  const float* A       = (const float*)d_in[1];
  const float* Wc      = (const float*)d_in[2];
  const float* bc      = (const float*)d_in[3];
  const float* gamma   = (const float*)d_in[4];
  const float* beta    = (const float*)d_in[5];
  const float* Wd      = (const float*)d_in[6];
  const float* bd      = (const float*)d_in[7];
  const float* gamma_d = (const float*)d_in[8];
  const float* beta_d  = (const float*)d_in[9];
  float* out = (float*)d_out;

  const long long total = (long long)N_BATCH * C_OUT * T_DIM * V_J; // 26,214,400

  // ws layout: Wb bf16 (64KB) | ybuf bf16 | rbuf bf16 | pbuf (1MB) | stats
  unsigned short* Wb   = (unsigned short*)d_ws;
  unsigned short* ybuf = Wb + 512 * 64;
  unsigned short* rbuf = ybuf + total;
  float* pbuf  = (float*)(rbuf + total);
  float* stats = pbuf + (long long)NBLK * 4 * C_OUT;

  w_pack_kernel<<<(512 * 64) / 256, 256, 0, stream>>>(Wc, Wd, Wb);

  dim3 grid1(T_DIM / TT, N_BATCH);   // 8 x 64 blocks
  gcn_wmma_kernel<<<grid1, 256, 0, stream>>>(x, A, Wb, bc, bd, ybuf, rbuf, pbuf);

  gcn_stats_kernel<<<2 * C_OUT, 256, 0, stream>>>(pbuf, stats);

  dim3 grid3((unsigned)(total / 4 / 256));
  gcn_finalize_kernel<<<grid3, 256, 0, stream>>>(ybuf, rbuf, stats,
                                                 gamma, beta, gamma_d, beta_d,
                                                 out, total);
}
